// AttentionLayer_88295937671814
// MI455X (gfx1250) — matrix operand order repaired and checked by reading, not compiled, not run on hardware
//
#include <hip/hip_runtime.h>
#include <hip/hip_bf16.h>
#include <math.h>

#define B_  32
#define S_  2048
#define H_  1024
#define E_  2048   // 2*H
#define W3_ 3072   // 3*H

typedef __attribute__((ext_vector_type(16))) __bf16 v16bf;
typedef __attribute__((ext_vector_type(8)))  float  v8f;

// ---------------- Kernel 1: hid_proj = hidden @ w_h^T + attn_b  (32x1024, tiny)
__global__ void __launch_bounds__(256)
k_hid_proj(const float* __restrict__ hidden,
           const float* __restrict__ attn_w,
           const float* __restrict__ attn_b,
           float* __restrict__ hid_proj) {
    __shared__ float hrow[H_];
    const int b = blockIdx.x >> 2;
    const int h = ((blockIdx.x & 3) << 8) + threadIdx.x;
    for (int i = threadIdx.x; i < H_; i += 256) hrow[i] = hidden[b * H_ + i];
    __syncthreads();
    const float4* wr = reinterpret_cast<const float4*>(attn_w + (size_t)h * W3_);
    float acc = 0.f;
    #pragma unroll 4
    for (int k = 0; k < H_ / 4; ++k) {
        float4 w4 = wr[k];
        acc += hrow[4*k+0]*w4.x + hrow[4*k+1]*w4.y + hrow[4*k+2]*w4.z + hrow[4*k+3]*w4.w;
    }
    hid_proj[b * H_ + h] = acc + attn_b[h];
}

// ---------------- Kernel 2: fused  enc @ w_e^T -> tanh -> dot(v) -> atomic scores
// grid: (512 row-tiles of 128, 8 h-chunks of 128), block 256 (8 waves)
#define MT 128
#define NT 128
#define KT 32
__global__ void __launch_bounds__(256)
k_scores(const float* __restrict__ enc,       // [B,S,E]
         const float* __restrict__ attn_w,    // [H,3H] ; w_e = attn_w[:, H:3H]
         const float* __restrict__ hid_proj,  // [B,H]
         const float* __restrict__ v_w,       // [H]
         float* __restrict__ scores)          // [B*S], pre-zeroed, atomic accum
{
    __shared__ __bf16 Als[MT * KT];   // 8 KB
    __shared__ __bf16 Bls[NT * KT];   // 8 KB

    const int tid  = threadIdx.x;
    const int lane = tid & 31;
    const int wave = tid >> 5;
    const int half = lane >> 4;      // 0: lanes 0-15, 1: lanes 16-31
    const int l16  = lane & 15;

    const int  tileM = blockIdx.x;             // 0..511
    const int  n0    = blockIdx.y * NT;        // h-chunk base
    const long m0    = (long)tileM * MT;       // global flattened (b,s) row
    const int  b     = (int)(m0 / S_);         // 128 | 2048 -> no straddle

    v8f acc[8] = {};                           // 8 N-tiles of 16x16 f32

    const size_t encBase = (size_t)m0 * E_;
    const int r    = tid >> 1;                 // row/col for cooperative loads
    const int kOff = (tid & 1) << 4;           // 0 or 16

    for (int kt = 0; kt < E_; kt += KT) {
        // ---- stage A tile: 128 rows x 32 k of encoder_outputs, f32 -> bf16
        {
            const float4* g = reinterpret_cast<const float4*>(
                enc + encBase + (size_t)r * E_ + kt + kOff);
            __bf16* dst = &Als[r * KT + kOff];
            #pragma unroll
            for (int i = 0; i < 4; ++i) {
                float4 v = g[i];
                dst[4*i+0] = (__bf16)v.x; dst[4*i+1] = (__bf16)v.y;
                dst[4*i+2] = (__bf16)v.z; dst[4*i+3] = (__bf16)v.w;
            }
        }
        // ---- stage B tile: 128 h x 32 k of w_e (contiguous in k), f32 -> bf16
        {
            const float4* g = reinterpret_cast<const float4*>(
                attn_w + (size_t)(n0 + r) * W3_ + H_ + kt + kOff);
            __bf16* dst = &Bls[r * KT + kOff];
            #pragma unroll
            for (int i = 0; i < 4; ++i) {
                float4 v = g[i];
                dst[4*i+0] = (__bf16)v.x; dst[4*i+1] = (__bf16)v.y;
                dst[4*i+2] = (__bf16)v.z; dst[4*i+3] = (__bf16)v.w;
            }
        }
        __syncthreads();

        // A fragment (16x32 bf16): lanes 0-15 K={0-7,16-23}, lanes 16-31 K={8-15,24-31}
        v16bf afrag;
        {
            const __bf16* arow = &Als[(wave * 16 + l16) * KT];
            #pragma unroll
            for (int i = 0; i < 16; ++i) {
                const int kk = (i < 8 ? i : i + 8) + half * 8;
                afrag[i] = arow[kk];
            }
        }
        // 8 WMMAs: B fragment (32x16 bf16): lanes 0-15 K={0-7,16-23}, lanes 16-31 K={8-15,24-31}
        #pragma unroll
        for (int t = 0; t < 8; ++t) {
            v16bf bfrag;
            const __bf16* brow = &Bls[(t * 16 + l16) * KT];
            #pragma unroll
            for (int i = 0; i < 16; ++i) bfrag[i] = brow[(i < 8 ? i : i + 8) + half * 8];
            acc[t] = __builtin_amdgcn_wmma_f32_16x16x32_bf16(
                false, afrag, false, bfrag, (short)0, acc[t], false, false);
        }
        __syncthreads();
    }

    // ---- epilogue: partial score = sum_h v[h] * tanh(hid_proj[h] + proj[h])
    float hp[8], vw[8];
    #pragma unroll
    for (int t = 0; t < 8; ++t) {
        const int n = n0 + t * 16 + l16;   // this lane's column per C layout
        hp[t] = hid_proj[b * H_ + n];
        vw[t] = v_w[n];
    }
    #pragma unroll
    for (int v = 0; v < 8; ++v) {
        float p = 0.f;
        #pragma unroll
        for (int t = 0; t < 8; ++t)
            p += vw[t] * tanhf(hp[t] + acc[t][v]);
        // reduce across the 16 lanes that share this output row
        #pragma unroll
        for (int m = 8; m >= 1; m >>= 1)
            p += __shfl_xor(p, m, 32);
        if (l16 == 0) {
            const long row = m0 + wave * 16 + half * 8 + v;   // C: M = v (+8 hi half)
            atomicAdd(&scores[row], p);
        }
    }
}

// ---------------- Kernel 3: softmax over S per batch row
__global__ void __launch_bounds__(256)
k_softmax(const float* __restrict__ scores, float* __restrict__ attn) {
    __shared__ float red[8];
    const int b = blockIdx.x, tid = threadIdx.x;
    const float* row = scores + (size_t)b * S_;
    float vals[8];
    float lmax = -3.0e38f;
    #pragma unroll
    for (int i = 0; i < 8; ++i) { vals[i] = row[tid + i * 256]; lmax = fmaxf(lmax, vals[i]); }
    #pragma unroll
    for (int m = 16; m >= 1; m >>= 1) lmax = fmaxf(lmax, __shfl_xor(lmax, m, 32));
    if ((tid & 31) == 0) red[tid >> 5] = lmax;
    __syncthreads();
    float bmax = red[0];
    #pragma unroll
    for (int i = 1; i < 8; ++i) bmax = fmaxf(bmax, red[i]);
    __syncthreads();
    float lsum = 0.f;
    #pragma unroll
    for (int i = 0; i < 8; ++i) { vals[i] = __expf(vals[i] - bmax); lsum += vals[i]; }
    #pragma unroll
    for (int m = 16; m >= 1; m >>= 1) lsum += __shfl_xor(lsum, m, 32);
    if ((tid & 31) == 0) red[tid >> 5] = lsum;
    __syncthreads();
    float bsum = 0.f;
    #pragma unroll
    for (int i = 0; i < 8; ++i) bsum += red[i];
    const float inv = 1.0f / bsum;
    float* out = attn + (size_t)b * S_;
    #pragma unroll
    for (int i = 0; i < 8; ++i) out[tid + i * 256] = vals[i] * inv;
}

// ---------------- Kernel 4: context[b,e] = sum_s w[b,s] * enc[b,s,e]  (512 MB stream)
__global__ void __launch_bounds__(256)
k_context(const float* __restrict__ enc, const float* __restrict__ attn,
          float* __restrict__ ctx) {
    __shared__ float w[S_];   // 8 KB
    const int b = blockIdx.y;
    const int e = blockIdx.x * 256 + threadIdx.x;
    for (int i = threadIdx.x; i < S_; i += 256) w[i] = attn[(size_t)b * S_ + i];
    __syncthreads();
    const float* base = enc + (size_t)b * S_ * E_ + e;
    float a0 = 0.f, a1 = 0.f, a2 = 0.f, a3 = 0.f;
    for (int s = 0; s < S_; s += 4) {
        a0 += w[s + 0] * base[(size_t)(s + 0) * E_];
        a1 += w[s + 1] * base[(size_t)(s + 1) * E_];
        a2 += w[s + 2] * base[(size_t)(s + 2) * E_];
        a3 += w[s + 3] * base[(size_t)(s + 3) * E_];
    }
    ctx[b * E_ + e] = (a0 + a1) + (a2 + a3);
}

extern "C" void kernel_launch(void* const* d_in, const int* in_sizes, int n_in,
                              void* d_out, int out_size, void* d_ws, size_t ws_size,
                              hipStream_t stream) {
    const float* hidden = (const float*)d_in[0];
    const float* enc    = (const float*)d_in[1];
    const float* attn_w = (const float*)d_in[2];
    const float* attn_b = (const float*)d_in[3];
    const float* v_w    = (const float*)d_in[4];

    float* ctx  = (float*)d_out;            // context: 32*2048
    float* attn = ctx + B_ * E_;            // attention_weights: 32*2048

    float* hid_proj = (float*)d_ws;         // 32*1024 f32
    float* scores   = hid_proj + B_ * H_;   // 32*2048 f32 (atomic-accumulated)

    hipMemsetAsync(scores, 0, (size_t)B_ * S_ * sizeof(float), stream);

    k_hid_proj<<<128, 256, 0, stream>>>(hidden, attn_w, attn_b, hid_proj);

    dim3 g2(S_ * B_ / MT, H_ / NT);         // 512 x 8
    k_scores<<<g2, 256, 0, stream>>>(enc, attn_w, hid_proj, v_w, scores);

    k_softmax<<<B_, 256, 0, stream>>>(scores, attn);

    dim3 g4(E_ / 256, B_);                  // 8 x 32
    k_context<<<g4, 256, 0, stream>>>(enc, attn, ctx);
}
